// GraphClassifier_88605175317070
// MI455X (gfx1250) — matrix operand order repaired and checked by reading, not compiled, not run on hardware
//
#include <hip/hip_runtime.h>
#include <hip/hip_bf16.h>

// ---------------------------------------------------------------------------
// Types / helpers
// ---------------------------------------------------------------------------
typedef __attribute__((ext_vector_type(16))) __bf16 v16bf;
typedef __attribute__((ext_vector_type(8)))  float  v8f;
typedef __attribute__((ext_vector_type(4)))  unsigned int u32x4;

union BF16x16 { v16bf v; u32x4 q[2]; unsigned short h[16]; };

__device__ __forceinline__ float bf2f(unsigned short u) {
    union { float f; unsigned int i; } v; v.i = ((unsigned int)u) << 16; return v.f;
}
__device__ __forceinline__ unsigned short f2bf(float f) {
    union { float f; unsigned int i; } v; v.f = f;
    unsigned int r = v.i + 0x7FFFu + ((v.i >> 16) & 1u);
    return (unsigned short)(r >> 16);
}
__device__ __forceinline__ float sigf(float x) { return 1.f / (1.f + __expf(-x)); }

// ---------------------------------------------------------------------------
// Elementwise / utility kernels
// ---------------------------------------------------------------------------
__global__ void k_cvt(unsigned short* __restrict__ dst, const float* __restrict__ src, size_t n) {
    size_t t = (size_t)blockIdx.x * blockDim.x + threadIdx.x;
    if (t < n) dst[t] = f2bf(src[t]);
}

// dst[C][R] = bf16(src[R][C])  (transpose)
__global__ void k_cvt_t(unsigned short* __restrict__ dst, const float* __restrict__ src, int R, int C) {
    int t = blockIdx.x * blockDim.x + threadIdx.x;
    if (t >= R * C) return;
    int r = t / C, c = t % C;
    dst[(size_t)c * R + r] = f2bf(src[t]);
}

__global__ void k_zero(float* __restrict__ p, size_t n) {
    size_t t = (size_t)blockIdx.x * blockDim.x + threadIdx.x;
    if (t < n) p[t] = 0.f;
}

// agg[dst[e]][h] += me[e][h]   (f32 atomics, H = 128)
__global__ void k_segsum(float* __restrict__ agg, const unsigned short* __restrict__ me,
                         const int* __restrict__ dsti, size_t n) {
    size_t t = (size_t)blockIdx.x * blockDim.x + threadIdx.x;
    if (t >= n) return;
    size_t e = t >> 7; int h = (int)(t & 127);
    unsafeAtomicAdd(&agg[(size_t)dsti[e] * 128 + h], bf2f(me[t]));
}

// dst = bf16(bf2f(mn) + agg)
__global__ void k_addn(unsigned short* __restrict__ dst, const unsigned short* __restrict__ mn,
                       const float* __restrict__ agg, size_t n) {
    size_t t = (size_t)blockIdx.x * blockDim.x + threadIdx.x;
    if (t < n) dst[t] = f2bf(bf2f(mn[t]) + agg[t]);
}

// dst = bf16(src * a[row])  (H = 128)
__global__ void k_scale(unsigned short* __restrict__ dst, const unsigned short* __restrict__ src,
                        const float* __restrict__ a, size_t n) {
    size_t t = (size_t)blockIdx.x * blockDim.x + threadIdx.x;
    if (t < n) dst[t] = f2bf(bf2f(src[t]) * a[t >> 7]);
}

// a[r] = sigmoid(sum_k T[r][k] * w2[k]),  K = 128
__global__ void k_rowdot_sig(float* __restrict__ a, const unsigned short* __restrict__ T,
                             const float* __restrict__ w2, int M) {
    int r = blockIdx.x * blockDim.x + threadIdx.x;
    if (r >= M) return;
    const unsigned short* row = T + (size_t)r * 128;
    float s = 0.f;
#pragma unroll 8
    for (int k = 0; k < 128; k++) s += bf2f(row[k]) * w2[k];
    a[r] = sigf(s);
}

// X = bf16(relu(node_h + gru_bias))   (H = 128)
__global__ void k_gruprep(unsigned short* __restrict__ X, const float* __restrict__ nh,
                          const float* __restrict__ bias, size_t n) {
    size_t t = (size_t)blockIdx.x * blockDim.x + threadIdx.x;
    if (t >= n) return;
    float v = nh[t] + bias[t & 127];
    X[t] = f2bf(fmaxf(v, 0.f));
}

// h0[b][h] = max_s node_h[b*256+s][h]
__global__ void k_h0max(float* __restrict__ h0, const float* __restrict__ nh) {
    int t = blockIdx.x * blockDim.x + threadIdx.x;  // 64*128 threads
    if (t >= 64 * 128) return;
    int b = t >> 7, h = t & 127;
    float m = -3.4e38f;
    for (int s = 0; s < 256; s++)
        m = fmaxf(m, nh[((size_t)b * 256 + s) * 128 + h]);
    h0[t] = m;
}

// ---------------------------------------------------------------------------
// WMMA GEMM:  C = act( [A0|A1|A2](gathered,scaled) @ W + bias + res )
//  block tile 128x128, 8 waves, each wave owns a 16-row slab and loops 8
//  16-col subtiles; K chunks of 32 (bf16), fp32 accumulate.
//  Gather index / row scale / A base pointer hoisted out of the K loop;
//  next-chunk A and W prefetched (global_prefetch_b8).
// ---------------------------------------------------------------------------
#define AS_STRIDE 40
#define BS_STRIDE 40

struct GemmP {
    const unsigned short* A0; const int* I0; int lda0; int K0;
    const unsigned short* A1; const int* I1; int lda1; int K1;
    const unsigned short* A2; const int* I2; int lda2; int K2;
    const unsigned short* W;  int ldw;
    const float* bias;
    const unsigned short* res; int ldr;     // bf16 residual or null
    const float* rowscale;                  // per-output-row A scale or null
    float* outf; unsigned short* outb; int ldo;
    int act;                                // 0 none, 1 relu
};

__global__ __launch_bounds__(256) void k_gemm(GemmP p) {
    __shared__ unsigned short As[128 * AS_STRIDE];
    __shared__ unsigned short Bs[128 * BS_STRIDE];

    const int m0 = blockIdx.x * 128;
    const int n0 = blockIdx.y * 128;
    const int t = threadIdx.x;
    const int wave = t >> 5, lane = t & 31;
    const int kh = lane >> 4, ln = lane & 15;

    // per-thread A-load coordinates (fixed for the whole kernel)
    const int rA = t >> 1, oA = (t & 1) * 16;
    const int growA = m0 + rA;
    const bool do_scale = (p.rowscale != nullptr);
    const float sc = do_scale ? p.rowscale[growA] : 1.f;
    // per-thread B-load coordinates
    const int kB = t >> 3, noB = (t & 7) * 16;

    v8f acc[8];
#pragma unroll
    for (int i = 0; i < 8; i++)
#pragma unroll
        for (int r = 0; r < 8; r++) acc[i][r] = 0.f;

    const unsigned short* Aseg[3] = { p.A0, p.A1, p.A2 };
    const int*            Iseg[3] = { p.I0, p.I1, p.I2 };
    int ldas[3] = { p.lda0, p.lda1, p.lda2 };
    int Ks[3]   = { p.K0,   p.K1,   p.K2   };

    int kw = 0;
    for (int s = 0; s < 3; s++) {
        int Kseg = Ks[s];
        if (Kseg == 0) continue;
        const int* idx = Iseg[s];
        // hoisted gather: row index + base pointer constant across the K loop
        long arow = idx ? (long)idx[growA] : (long)growA;
        const unsigned short* abase = Aseg[s] + (size_t)arow * ldas[s] + oA;
        const unsigned short* wbase = p.W + (size_t)(kw + kB) * p.ldw + n0 + noB;

        for (int kc = 0; kc < Kseg; kc += 32) {
            // speculative prefetch of next chunk (global_prefetch_b8)
            if (kc + 32 < Kseg) {
                __builtin_prefetch(abase + kc + 32, 0, 0);
                __builtin_prefetch(wbase + (size_t)(kc + 32) * p.ldw, 0, 0);
            }
            // --- A tile: 128 rows x 32 halves, row-major in LDS
            {
                const unsigned short* src = abase + kc;
                unsigned short* dst = &As[rA * AS_STRIDE + oA];
                if (do_scale) {
#pragma unroll
                    for (int i = 0; i < 16; i++) dst[i] = f2bf(bf2f(src[i]) * sc);
                } else {
                    *(u32x4*)(dst)     = *(const u32x4*)(src);
                    *(u32x4*)(dst + 8) = *(const u32x4*)(src + 8);
                }
            }
            // --- B tile: transpose W[k][n] -> Bs[n][k] (coalesced global read)
            {
                const unsigned short* src = wbase + (size_t)kc * p.ldw;
                BF16x16 tmp;
                tmp.q[0] = *(const u32x4*)src;
                tmp.q[1] = *(const u32x4*)(src + 8);
#pragma unroll
                for (int i = 0; i < 16; i++) Bs[(noB + i) * BS_STRIDE + kB] = tmp.h[i];
            }
            __syncthreads();
            // --- fragments + WMMA
            int am = wave * 16 + ln;
            BF16x16 af;
            af.q[0] = *(const u32x4*)&As[am * AS_STRIDE + 16 * kh];
            af.q[1] = *(const u32x4*)&As[am * AS_STRIDE + 16 * kh + 8];
#pragma unroll
            for (int nt = 0; nt < 8; nt++) {
                int bn = nt * 16 + ln;
                BF16x16 bf;
                bf.q[0] = *(const u32x4*)&Bs[bn * BS_STRIDE + 16 * kh];
                bf.q[1] = *(const u32x4*)&Bs[bn * BS_STRIDE + 16 * kh + 8];
                acc[nt] = __builtin_amdgcn_wmma_f32_16x16x32_bf16(
                    false, af.v, false, bf.v, (short)0, acc[nt], false, false);
            }
            __syncthreads();
        }
        kw += Kseg;
    }
    // --- epilogue
#pragma unroll
    for (int nt = 0; nt < 8; nt++) {
#pragma unroll
        for (int r = 0; r < 8; r++) {
            int row = m0 + wave * 16 + r + 8 * kh;
            int col = n0 + nt * 16 + ln;
            float v = acc[nt][r];
            if (p.bias) v += p.bias[col];
            if (p.res)  v += bf2f(p.res[(size_t)row * p.ldr + col]);
            if (p.act == 1) v = fmaxf(v, 0.f);
            if (p.outf) p.outf[(size_t)row * p.ldo + col] = v;
            if (p.outb) p.outb[(size_t)row * p.ldo + col] = f2bf(v);
        }
    }
}

// ---------------------------------------------------------------------------
// Persistent bidirectional GRU: blockIdx.x = direction. 256 threads, 8 waves.
// h (64x128) bf16 in LDS; per step: gh = h @ Whh^T via WMMA (96 tiles), gates
// in f32 registers, h updated, output written to gru_out[node][d*128 + h].
// ---------------------------------------------------------------------------
__global__ __launch_bounds__(256) void k_gru(
    unsigned short* __restrict__ out,      // [N][256] bf16 (fwd cols 0..127, bwd 128..255)
    const float* __restrict__ gi_all,      // [2][N][384] = x @ Wih^T + bih
    const unsigned short* __restrict__ whh,// [2][384][128] bf16 (native Whh layout)
    const float* __restrict__ bhh,         // [2][384]
    const float* __restrict__ h0)          // [64][128]
{
    __shared__ unsigned short hs[64 * 128];
    __shared__ unsigned short ghs[64 * 384];

    const int d = blockIdx.x;
    const int t = threadIdx.x;
    const int wave = t >> 5, lane = t & 31;
    const int kh = lane >> 4, ln = lane & 15;

    const float* gi_d = gi_all + (size_t)d * 16384 * 384;
    const unsigned short* whh_d = whh + (size_t)d * 384 * 128;
    const float* bhh_d = bhh + d * 384;

    const int m  = t >> 2;            // batch row 0..63 owned for gate math
    const int c0 = (t & 3) * 32;      // 32 contiguous cols per thread
    float hreg[32];
#pragma unroll
    for (int i = 0; i < 32; i++) {
        float v = h0[m * 128 + c0 + i];
        hreg[i] = v;
        hs[m * 128 + c0 + i] = f2bf(v);
    }
    __syncthreads();

    const int mt = wave >> 1;             // m-tile 0..3
    const int ntbase = (wave & 1) * 12;   // 12 n-tiles per wave (24 total = 384 cols)

    for (int s = 0; s < 256; s++) {
        int sin = d ? (255 - s) : s;
        // gh = h @ Whh^T (WMMA)
        for (int j = 0; j < 12; j++) {
            int nt = ntbase + j;
            v8f acc;
#pragma unroll
            for (int r = 0; r < 8; r++) acc[r] = 0.f;
#pragma unroll
            for (int kc = 0; kc < 4; kc++) {
                int am = mt * 16 + ln;
                BF16x16 af;
                af.q[0] = *(const u32x4*)&hs[am * 128 + kc * 32 + 16 * kh];
                af.q[1] = *(const u32x4*)&hs[am * 128 + kc * 32 + 16 * kh + 8];
                int bn = nt * 16 + ln;
                const unsigned short* wp = whh_d + (size_t)bn * 128 + kc * 32 + 16 * kh;
                BF16x16 bf;
                bf.q[0] = *(const u32x4*)wp;
                bf.q[1] = *(const u32x4*)(wp + 8);
                acc = __builtin_amdgcn_wmma_f32_16x16x32_bf16(
                    false, af.v, false, bf.v, (short)0, acc, false, false);
            }
#pragma unroll
            for (int r = 0; r < 8; r++) {
                int row = mt * 16 + r + 8 * kh;
                int col = nt * 16 + ln;
                ghs[row * 384 + col] = f2bf(acc[r]);
            }
        }
        __syncthreads();
        // gates
        const float* gi = gi_d + (size_t)(m * 256 + sin) * 384;
        unsigned short* orow = out + (size_t)(m * 256 + sin) * 256 + d * 128;
#pragma unroll 4
        for (int i = 0; i < 32; i++) {
            int c = c0 + i;
            float hr = bf2f(ghs[m * 384 + c])       + bhh_d[c];
            float hz = bf2f(ghs[m * 384 + c + 128]) + bhh_d[c + 128];
            float hn = bf2f(ghs[m * 384 + c + 256]) + bhh_d[c + 256];
            float r = sigf(gi[c] + hr);
            float z = sigf(gi[c + 128] + hz);
            float n = tanhf(gi[c + 256] + r * hn);
            float hnew = (1.f - z) * n + z * hreg[i];
            hreg[i] = hnew;
            hs[m * 128 + c] = f2bf(hnew);
            orow[c] = f2bf(hnew);
        }
        __syncthreads();
    }
}

// ---------------------------------------------------------------------------
// Readout: conv = tanh(h2[head] + rel_emb[trel] - h2[tail]); out = lin2(lin1(conv))
// ---------------------------------------------------------------------------
__global__ void k_readout(float* __restrict__ out, const unsigned short* __restrict__ nh2,
                          const float* __restrict__ rel_emb, const int* __restrict__ trel,
                          const int* __restrict__ head, const int* __restrict__ tail,
                          const float* __restrict__ l1w, const float* __restrict__ l1b,
                          const float* __restrict__ l2w, const float* __restrict__ l2b) {
    int b = threadIdx.x;
    if (b >= 64) return;
    const unsigned short* hr = nh2 + (size_t)head[b] * 128;
    const unsigned short* tr = nh2 + (size_t)tail[b] * 128;
    const float* rr = rel_emb + (size_t)trel[b] * 128;
    float conv[128];
    for (int h = 0; h < 128; h++)
        conv[h] = tanhf(bf2f(hr[h]) + rr[h] - bf2f(tr[h]));
    float o = l2b[0];
    for (int j = 0; j < 16; j++) {
        float tj = l1b[j];
        for (int h = 0; h < 128; h++) tj += conv[h] * l1w[h * 16 + j];
        o += tj * l2w[j];
    }
    out[b] = o;
}

// ---------------------------------------------------------------------------
// Host orchestration
// ---------------------------------------------------------------------------
extern "C" void kernel_launch(void* const* d_in, const int* in_sizes, int n_in,
                              void* d_out, int out_size, void* d_ws, size_t ws_size,
                              hipStream_t stream) {
    const int N = 16384, E = 262144, H = 128, INP = 64, R = 32;

    const float* node_feat  = (const float*)d_in[0];
    const float* rel_emb    = (const float*)d_in[1];
    const float* W_i_node   = (const float*)d_in[2];
    const float* W_i_edge   = (const float*)d_in[3];
    const float* W_att_in1  = (const float*)d_in[4];
    const float* W_att_in2  = (const float*)d_in[5];
    const float* W_h_node   = (const float*)d_in[6];
    const float* W_h_edge   = (const float*)d_in[7];
    const float* W_att1     = (const float*)d_in[8];
    const float* W_att2     = (const float*)d_in[9];
    const float* comm_W     = (const float*)d_in[10];
    const float* W_o        = (const float*)d_in[11];
    const float* b_o        = (const float*)d_in[12];
    const float* gru_bias   = (const float*)d_in[13];
    const float* gru_Wih    = (const float*)d_in[14];
    const float* gru_Whh    = (const float*)d_in[15];
    const float* gru_bih    = (const float*)d_in[16];
    const float* gru_bhh    = (const float*)d_in[17];
    const float* lin1_W     = (const float*)d_in[18];
    const float* lin1_b     = (const float*)d_in[19];
    const float* lin2_W     = (const float*)d_in[20];
    const float* lin2_b     = (const float*)d_in[21];
    const int* src_idx      = (const int*)d_in[22];
    const int* dst_idx      = (const int*)d_in[23];
    const int* edge_type    = (const int*)d_in[24];
    const int* edge_label   = (const int*)d_in[25];
    const int* trel_idx     = (const int*)d_in[26];
    const int* head_idx     = (const int*)d_in[27];
    const int* tail_idx     = (const int*)d_in[28];

    size_t off = 0;
    auto alloc = [&](size_t bytes) -> void* {
        void* p = (char*)d_ws + off;
        off += (bytes + 255) & ~(size_t)255;
        return p;
    };
    unsigned short* nf_b         = (unsigned short*)alloc((size_t)N * INP * 2);
    unsigned short* rel_b        = (unsigned short*)alloc((size_t)R * H * 2);
    unsigned short* w_i_node_b   = (unsigned short*)alloc(64 * 128 * 2);
    unsigned short* w_i_edge_b   = (unsigned short*)alloc(256 * 128 * 2);
    unsigned short* w_att_in1_b  = (unsigned short*)alloc(256 * 128 * 2);
    unsigned short* w_h_node_b   = (unsigned short*)alloc(2 * 128 * 128 * 2);
    unsigned short* w_h_edge_b   = (unsigned short*)alloc(2 * 128 * 128 * 2);
    unsigned short* w_att1_b     = (unsigned short*)alloc(2 * 256 * 128 * 2);
    unsigned short* comm_b       = (unsigned short*)alloc(384 * 128 * 2);
    unsigned short* w_o_b        = (unsigned short*)alloc(256 * 128 * 2);
    unsigned short* wih_t_b      = (unsigned short*)alloc(2 * 128 * 384 * 2);
    unsigned short* whh_b        = (unsigned short*)alloc(2 * 384 * 128 * 2);
    unsigned short* input_node_b = (unsigned short*)alloc((size_t)N * H * 2);
    unsigned short* mnA_b        = (unsigned short*)alloc((size_t)N * H * 2);
    unsigned short* mnB_b        = (unsigned short*)alloc((size_t)N * H * 2);
    unsigned short* nodeA_b      = (unsigned short*)alloc((size_t)N * H * 2);
    unsigned short* agg_b        = (unsigned short*)alloc((size_t)N * H * 2);
    unsigned short* input_edge_b = (unsigned short*)alloc((size_t)E * H * 2);
    unsigned short* me_b         = (unsigned short*)alloc((size_t)E * H * 2);
    unsigned short* T_b          = (unsigned short*)alloc((size_t)E * H * 2);
    float*          a_f          = (float*)alloc((size_t)E * 4);
    float*          agg_f        = (float*)alloc((size_t)N * H * 4);
    float*          node_h_f     = (float*)alloc((size_t)N * H * 4);
    unsigned short* X_b          = (unsigned short*)alloc((size_t)N * H * 2);
    float*          gi_f         = (float*)alloc((size_t)2 * N * 384 * 4);
    float*          h0_f         = (float*)alloc(64 * 128 * 4);
    unsigned short* gruout_b     = (unsigned short*)alloc((size_t)N * 256 * 2);
    unsigned short* nh2_b        = (unsigned short*)alloc((size_t)N * H * 2);

    auto cvt = [&](unsigned short* dst, const float* src, size_t n) {
        k_cvt<<<dim3((unsigned)((n + 255) / 256)), dim3(256), 0, stream>>>(dst, src, n);
    };
    auto gemm = [&](int M, int Nc,
                    const unsigned short* A0, const int* I0, int lda0, int K0,
                    const unsigned short* A1, const int* I1, int lda1, int K1,
                    const unsigned short* A2, const int* I2, int lda2, int K2,
                    const unsigned short* W, int ldw,
                    const float* bias, const unsigned short* res, int ldr,
                    const float* rowscale,
                    float* outf, unsigned short* outb, int ldo, int act) {
        GemmP p{ A0, I0, lda0, K0, A1, I1, lda1, K1, A2, I2, lda2, K2,
                 W, ldw, bias, res, ldr, rowscale, outf, outb, ldo, act };
        k_gemm<<<dim3(M / 128, Nc / 128), dim3(256), 0, stream>>>(p);
    };

    // ---- bf16 conversions (params + node features)
    cvt(nf_b, node_feat, (size_t)N * INP);
    cvt(rel_b, rel_emb, (size_t)R * H);
    cvt(w_i_node_b, W_i_node, 64 * 128);
    cvt(w_i_edge_b, W_i_edge, 256 * 128);
    cvt(w_att_in1_b, W_att_in1, 256 * 128);
    cvt(w_h_node_b, W_h_node, 2 * 128 * 128);
    cvt(w_h_edge_b, W_h_edge, 2 * 128 * 128);
    cvt(w_att1_b, W_att1, 2 * 256 * 128);
    cvt(comm_b, comm_W, 384 * 128);
    cvt(w_o_b, W_o, 256 * 128);
    cvt(whh_b, gru_Whh, 2 * 384 * 128);
    for (int d = 0; d < 2; d++)
        k_cvt_t<<<dim3((384 * 128 + 255) / 256), dim3(256), 0, stream>>>(
            wih_t_b + (size_t)d * 128 * 384, gru_Wih + (size_t)d * 384 * 128, 384, 128);

    // ---- input_node = relu(node_feat @ W_i_node)
    gemm(N, 128, nf_b, nullptr, 64, 64, nullptr, nullptr, 0, 0, nullptr, nullptr, 0, 0,
         w_i_node_b, 128, nullptr, nullptr, 0, nullptr, nullptr, input_node_b, 128, 1);

    // ---- input_edge = relu([nf[src] | rel[type] | nf[dst]] @ W_i_edge)
    gemm(E, 128, nf_b, src_idx, 64, 64, rel_b, edge_type, 128, 128, nf_b, dst_idx, 64, 64,
         w_i_edge_b, 128, nullptr, nullptr, 0, nullptr, nullptr, input_edge_b, 128, 1);

    // ---- input attention: T = relu([mn[src] | input_edge] @ W_att_in1); a = sig(T·w2)
    gemm(E, 128, input_node_b, src_idx, 128, 128, input_edge_b, nullptr, 128, 128,
         nullptr, nullptr, 0, 0, w_att_in1_b, 128, nullptr, nullptr, 0, nullptr,
         nullptr, T_b, 128, 1);
    k_rowdot_sig<<<dim3((E + 255) / 256), dim3(256), 0, stream>>>(a_f, T_b, W_att_in2, E);
    k_scale<<<dim3((unsigned)(((size_t)E * H) / 256)), dim3(256), 0, stream>>>(
        me_b, input_edge_b, a_f, (size_t)E * H);

    // ---- message passing layers
    const unsigned short* mn_prev = input_node_b;
    unsigned short* mn_cur = mnA_b;
    size_t nEH = (size_t)E * H, nNH = (size_t)N * H;
    for (int d = 0; d < 2; d++) {
        k_zero<<<dim3((unsigned)(nNH / 256)), dim3(256), 0, stream>>>(agg_f, nNH);
        k_segsum<<<dim3((unsigned)(nEH / 256)), dim3(256), 0, stream>>>(agg_f, me_b, dst_idx, nEH);
        k_addn<<<dim3((unsigned)(nNH / 256)), dim3(256), 0, stream>>>(nodeA_b, mn_prev, agg_f, nNH);
        // message_node = relu((mn + agg) @ W_h_node[d])
        gemm(N, 128, nodeA_b, nullptr, 128, 128, nullptr, nullptr, 0, 0, nullptr, nullptr, 0, 0,
             w_h_node_b + (size_t)d * 128 * 128, 128, nullptr, nullptr, 0, nullptr,
             nullptr, mn_cur, 128, 1);
        // a = sig(relu([me | rel[label]] @ W_att1[d]) · W_att2[d])
        gemm(E, 128, me_b, nullptr, 128, 128, rel_b, edge_label, 128, 128,
             nullptr, nullptr, 0, 0, w_att1_b + (size_t)d * 256 * 128, 128,
             nullptr, nullptr, 0, nullptr, nullptr, T_b, 128, 1);
        k_rowdot_sig<<<dim3((E + 255) / 256), dim3(256), 0, stream>>>(a_f, T_b, W_att2 + d * 128, E);
        // me = relu(input_edge + (mn[src]*a) @ W_h_edge[d])
        gemm(E, 128, mn_cur, src_idx, 128, 128, nullptr, nullptr, 0, 0, nullptr, nullptr, 0, 0,
             w_h_edge_b + (size_t)d * 128 * 128, 128, nullptr, input_edge_b, 128, a_f,
             nullptr, me_b, 128, 1);
        mn_prev = mn_cur;
        mn_cur = mnB_b;
    }

    // ---- final aggregation + comm
    k_zero<<<dim3((unsigned)(nNH / 256)), dim3(256), 0, stream>>>(agg_f, nNH);
    k_segsum<<<dim3((unsigned)(nEH / 256)), dim3(256), 0, stream>>>(agg_f, me_b, dst_idx, nEH);
    cvt(agg_b, agg_f, nNH);
    gemm(N, 128, agg_b, nullptr, 128, 128, mn_prev, nullptr, 128, 128,
         input_node_b, nullptr, 128, 128, comm_b, 128, nullptr, nullptr, 0, nullptr,
         node_h_f, nullptr, 128, 0);

    // ---- GRU
    k_gruprep<<<dim3((unsigned)(nNH / 256)), dim3(256), 0, stream>>>(X_b, node_h_f, gru_bias, nNH);
    k_h0max<<<dim3(32), dim3(256), 0, stream>>>(h0_f, node_h_f);
    for (int d = 0; d < 2; d++)
        gemm(N, 384, X_b, nullptr, 128, 128, nullptr, nullptr, 0, 0, nullptr, nullptr, 0, 0,
             wih_t_b + (size_t)d * 128 * 384, 384, gru_bih + d * 384, nullptr, 0, nullptr,
             gi_f + (size_t)d * N * 384, nullptr, 384, 0);
    k_gru<<<dim3(2), dim3(256), 0, stream>>>(gruout_b, gi_f, whh_b, gru_bhh, h0_f);

    // ---- node_h2 = relu(gru_out @ W_o + b_o)
    gemm(N, 128, gruout_b, nullptr, 256, 256, nullptr, nullptr, 0, 0, nullptr, nullptr, 0, 0,
         w_o_b, 128, b_o, nullptr, 0, nullptr, nullptr, nh2_b, 128, 1);

    // ---- readout
    k_readout<<<dim3(1), dim3(64), 0, stream>>>((float*)d_out, nh2_b, rel_emb, trel_idx,
                                                head_idx, tail_idx, lin1_W, lin1_b,
                                                lin2_W, lin2_b);
    (void)in_sizes; (void)n_in; (void)out_size; (void)ws_size;
}
